// GraphAttentionLayer_20117626815129
// MI455X (gfx1250) — matrix operand order repaired and checked by reading, not compiled, not run on hardware
//
#include <hip/hip_runtime.h>
#include <hip/hip_bf16.h>

// ---------------------------------------------------------------------------
// GAT layer for MI455X (gfx1250, wave32).
// Dominant cost: per-edge MLP (T=261632) x (384->128->384) = 51.4 GFLOP via
// V_WMMA_F32_16X16X32_BF16 (f32 accumulate).
// v2: weights pre-packed in WMMA-fragment order, staged into LDS with
// GLOBAL_LOAD_ASYNC_TO_LDS_B128 (ASYNCcnt), W2 overlays the cur_t tile after
// layer 1; two n-tiles per A-fragment for 2x WMMA ILP.
// Workspace (~0.97 MB): Wh[64K] f[512] g[512] hp[64K] pooled[64K] counts[512]
//                       bpack1 bf16[49152], bpack2 bf16[49152]
// ---------------------------------------------------------------------------

typedef __attribute__((ext_vector_type(16))) __bf16 v16bf;
typedef __attribute__((ext_vector_type(8)))  float  v8f;

#define O_N   512
#define D     128
#define CT    384
#define CTP   392           // As LDS stride (halves): 784B = 49*16, 196w ≡ 4 mod 64 banks
#define HPAD  136           // Hs LDS stride (halves): 272B = 17*16,  68w ≡ 4 mod 64 banks
#define TM    128
#define T_EDGES 261632      // 512*511

// LDS carve (bytes): [0,100352) As | Ws2 overlay ; [100352,135168) Hs ;
// [135168,233472) Ws1 ; [233472,234496) Es/Eo
#define SM_AS   0
#define SM_HS   100352
#define SM_WS1  135168
#define SM_ES   233472
#define SM_SIZE 234496

static __device__ inline v16bf ldfrag(const __bf16* p0, const __bf16* p1) {
    union { uint4 q[2]; v16bf v; } u;
    u.q[0] = *reinterpret_cast<const uint4*>(p0);
    u.q[1] = *reinterpret_cast<const uint4*>(p1);
    return u.v;
}

// Async byte copy global->LDS, nbytes must be multiple of 16*256.
static __device__ inline void async_copy_b128(unsigned lds_base, const void* gsrc,
                                              int nbytes, int t) {
    for (int i = t * 16; i < nbytes; i += 256 * 16) {
        unsigned lo = lds_base + i;
        int go = i;
        asm volatile("global_load_async_to_lds_b128 %0, %1, %2"
                     :: "v"(lo), "v"(go), "s"(gsrc) : "memory");
    }
}

// --------- pack weights f32 -> bf16 in WMMA B-fragment order ----------------
// bpack1: 8 ntiles x 12 ktiles (W1 384x128); bpack2: 24 ntiles x 4 ktiles (W2 128x384)
__global__ void k_prep(const float* __restrict__ W1, const float* __restrict__ W2,
                       __bf16* __restrict__ bpack1, __bf16* __restrict__ bpack2) {
    int t = blockIdx.x * 256 + threadIdx.x;          // 98304 total
    int e    = t & 15;
    int lane = (t >> 4) & 31;
    int fk   = (t >> 9) & 95;                        // fragment id within its pack
    int khalf = (lane >> 4) << 4;                    // 0 or 16
    int ncol  = lane & 15;
    if (t < 128 * 384) {
        int kt = fk % 12, nt = fk / 12;
        int K = kt * 32 + khalf + e;                 // < 384
        int n = nt * 16 + ncol;                      // < 128
        bpack1[t] = (__bf16)W1[K * 128 + n];
    } else {
        int kt = fk & 3, nt = fk >> 2;
        int K = kt * 32 + khalf + e;                 // < 128
        int n = nt * 16 + ncol;                      // < 384
        bpack2[t - 128 * 384] = (__bf16)W2[K * 384 + n];
    }
}

// ------------------- Wh = obj@W ; f = Wh@a1 ; g = Wh@a2 ---------------------
__global__ __launch_bounds__(128) void k_wh(const float* __restrict__ obj,
                                            const float* __restrict__ W,
                                            const float* __restrict__ a,
                                            float* __restrict__ Wh,
                                            float* __restrict__ f,
                                            float* __restrict__ g) {
    __shared__ float xo[128], r1[128], r2[128];
    int i = blockIdx.x, d = threadIdx.x;
    xo[d] = obj[i * 128 + d];
    __syncthreads();
    float acc = 0.f;
    for (int k = 0; k < 128; ++k) acc = fmaf(xo[k], W[k * 128 + d], acc);
    Wh[i * 128 + d] = acc;
    r1[d] = acc * a[d];
    r2[d] = acc * a[128 + d];
    __syncthreads();
    for (int s = 64; s > 0; s >>= 1) {
        if (d < s) { r1[d] += r1[d + s]; r2[d] += r2[d + s]; }
        __syncthreads();
    }
    if (d == 0) { f[i] = r1[0]; g[i] = r2[0]; }
}

// ------------- attention softmax (diag masked) + h_prime = relu(att@Wh) -----
__global__ __launch_bounds__(128) void k_att(const float* __restrict__ Wh,
                                             const float* __restrict__ f,
                                             const float* __restrict__ g,
                                             float* __restrict__ hp) {
    __shared__ float le[O_N];
    __shared__ float red[128];
    int i = blockIdx.x, t = threadIdx.x;
    float fi = f[i];
    float mx = -3.0e38f;
    for (int j = t; j < O_N; j += 128) {
        float e;
        if (j == i) e = -3.0e38f;
        else { float x = fi + g[j]; e = x > 0.f ? x : 0.2f * x; }
        le[j] = e;
        mx = fmaxf(mx, e);
    }
    red[t] = mx; __syncthreads();
    for (int s = 64; s > 0; s >>= 1) { if (t < s) red[t] = fmaxf(red[t], red[t + s]); __syncthreads(); }
    mx = red[0]; __syncthreads();
    float sm = 0.f;
    for (int j = t; j < O_N; j += 128) {
        float p = (j == i) ? 0.f : __expf(le[j] - mx);
        le[j] = p; sm += p;
    }
    red[t] = sm; __syncthreads();
    for (int s = 64; s > 0; s >>= 1) { if (t < s) red[t] += red[t + s]; __syncthreads(); }
    float inv = 1.f / red[0];
    float acc = 0.f;
    for (int j = 0; j < O_N; ++j) acc = fmaf(le[j], Wh[j * 128 + t], acc);
    float v = acc * inv;
    hp[i * 128 + t] = v > 0.f ? v : 0.f;
}

__global__ void k_zero(float* __restrict__ p, int n) {
    int i = blockIdx.x * 256 + threadIdx.x;
    if (i < n) p[i] = 0.f;
}

__global__ void k_counts(const int* __restrict__ edges, float* __restrict__ counts, int T) {
    int e = blockIdx.x * 256 + threadIdx.x;
    if (e < T) {
        atomicAdd(&counts[edges[2 * e]],     1.f);
        atomicAdd(&counts[edges[2 * e + 1]], 1.f);
    }
}

// ------------------- the WMMA edge MLP --------------------------------------
__global__ __launch_bounds__(256) void k_edge(const float* __restrict__ hp,
                                              const float* __restrict__ pred,
                                              const int* __restrict__ edges,
                                              const __bf16* __restrict__ bpack1,
                                              const float* __restrict__ b1,
                                              const __bf16* __restrict__ bpack2,
                                              const float* __restrict__ b2,
                                              float* __restrict__ pooled,
                                              float* __restrict__ out_p) {
    __shared__ __align__(16) unsigned char smem[SM_SIZE];
    __bf16* As  = (__bf16*)(smem + SM_AS);    // [128][CTP] cur_t tile
    __bf16* Ws2 = (__bf16*)(smem + SM_AS);    // overlays As after layer 1
    __bf16* Hs  = (__bf16*)(smem + SM_HS);    // [128][HPAD] layer-1 activations
    __bf16* Ws1 = (__bf16*)(smem + SM_WS1);   // fragment-packed W1 (96KB)
    int*    Es  = (int*)(smem + SM_ES);
    int*    Eo  = Es + TM;

    int t = threadIdx.x;
    long e0 = (long)blockIdx.x * TM;

    if (t < TM) {
        Es[t] = edges[2 * (e0 + t)];
        Eo[t] = edges[2 * (e0 + t) + 1];
    }
    // start the layer-1 weight DMA while we build the activation tile
    async_copy_b128((unsigned)(unsigned long long)(uintptr_t)(void*)Ws1,
                    (const void*)bpack1, 128 * 384 * 2, t);
    __syncthreads();

    // cur_t = [hp[s] | pred | hp[o]] as bf16 in LDS
    for (int gi = t; gi < TM * 96; gi += 256) {
        int lr = gi / 96;
        int c4 = (gi - lr * 96) * 4;
        const float* src;
        if (c4 < 128)      src = hp + (long)Es[lr] * 128 + c4;
        else if (c4 < 256) src = pred + (e0 + lr) * 128 + (c4 - 128);
        else               src = hp + (long)Eo[lr] * 128 + (c4 - 256);
        float4 v = *reinterpret_cast<const float4*>(src);
        union { __bf16 h[4]; uint2 q; } u;
        u.h[0] = (__bf16)v.x; u.h[1] = (__bf16)v.y;
        u.h[2] = (__bf16)v.z; u.h[3] = (__bf16)v.w;
        *reinterpret_cast<uint2*>(&As[lr * CTP + c4]) = u.q;
    }
    asm volatile("s_wait_asynccnt 0" ::: "memory");
    __syncthreads();

    int lane    = t & 31;
    int wave    = t >> 5;
    int m0      = wave * 16;
    int halfSel = (lane >> 4) & 1;
    int arow    = m0 + (lane & 15);
    int col16   = lane & 15;
    int mrow    = m0 + halfSel * 8;

    // ---- layer 1: h1 = relu(cur_t @ W1 + b1), two n-tiles per A fragment ----
    for (int ntp = 0; ntp < 4; ++ntp) {
        int nA = (2 * ntp) * 16 + col16;
        int nB = (2 * ntp + 1) * 16 + col16;
        float bvA = b1[nA], bvB = b1[nB];
        v8f c0 = { bvA, bvA, bvA, bvA, bvA, bvA, bvA, bvA };
        v8f c1 = { bvB, bvB, bvB, bvB, bvB, bvB, bvB, bvB };
        for (int kt = 0; kt < 12; ++kt) {
            int ab = kt * 32 + halfSel * 16;
            v16bf a = ldfrag(&As[arow * CTP + ab], &As[arow * CTP + ab + 8]);
            int f0 = (((2 * ntp) * 12 + kt) * 32 + lane) * 16;
            int f1 = (((2 * ntp + 1) * 12 + kt) * 32 + lane) * 16;
            v16bf b0 = ldfrag(Ws1 + f0, Ws1 + f0 + 8);
            v16bf bb = ldfrag(Ws1 + f1, Ws1 + f1 + 8);
            c0 = __builtin_amdgcn_wmma_f32_16x16x32_bf16(false, a, false, b0,
                                                         (short)0, c0, false, false);
            c1 = __builtin_amdgcn_wmma_f32_16x16x32_bf16(false, a, false, bb,
                                                         (short)0, c1, false, false);
        }
#pragma unroll
        for (int r = 0; r < 8; ++r) {
            float v0 = c0[r] > 0.f ? c0[r] : 0.f;
            float v1 = c1[r] > 0.f ? c1[r] : 0.f;
            Hs[(mrow + r) * HPAD + (2 * ntp) * 16 + col16]     = (__bf16)v0;
            Hs[(mrow + r) * HPAD + (2 * ntp + 1) * 16 + col16] = (__bf16)v1;
        }
    }
    __syncthreads();                       // As is now dead everywhere

    // DMA layer-2 weights into the As region
    async_copy_b128((unsigned)(unsigned long long)(uintptr_t)(void*)Ws2,
                    (const void*)bpack2, 384 * 128 * 2, t);
    asm volatile("s_wait_asynccnt 0" ::: "memory");
    __syncthreads();

    // ---- layer 2: new_t = relu(h1 @ W2 + b2), two n-tiles per A fragment ----
    for (int ntp = 0; ntp < 12; ++ntp) {
        int nt0 = 2 * ntp, nt1 = 2 * ntp + 1;
        int nA = nt0 * 16 + col16;
        int nB = nt1 * 16 + col16;
        float bvA = b2[nA], bvB = b2[nB];
        v8f c0 = { bvA, bvA, bvA, bvA, bvA, bvA, bvA, bvA };
        v8f c1 = { bvB, bvB, bvB, bvB, bvB, bvB, bvB, bvB };
        for (int kt = 0; kt < 4; ++kt) {
            int ab = kt * 32 + halfSel * 16;
            v16bf a = ldfrag(&Hs[arow * HPAD + ab], &Hs[arow * HPAD + ab + 8]);
            int f0 = ((nt0 * 4 + kt) * 32 + lane) * 16;
            int f1 = ((nt1 * 4 + kt) * 32 + lane) * 16;
            v16bf b0 = ldfrag(Ws2 + f0, Ws2 + f0 + 8);
            v16bf bb = ldfrag(Ws2 + f1, Ws2 + f1 + 8);
            c0 = __builtin_amdgcn_wmma_f32_16x16x32_bf16(false, a, false, b0,
                                                         (short)0, c0, false, false);
            c1 = __builtin_amdgcn_wmma_f32_16x16x32_bf16(false, a, false, bb,
                                                         (short)0, c1, false, false);
        }
        // scatter both tiles
#pragma unroll
        for (int half = 0; half < 2; ++half) {
            int nt2 = 2 * ntp + half;
            int n2 = nt2 * 16 + col16;
            v8f& c = half ? c1 : c0;
            if (nt2 < 8) {                               // new_s -> pooled[s]
#pragma unroll
                for (int r = 0; r < 8; ++r) {
                    float v = fmaxf(c[r], 0.f);
                    atomicAdd(&pooled[(long)Es[mrow + r] * 128 + n2], v);
                }
            } else if (nt2 < 16) {                       // new_p -> output
#pragma unroll
                for (int r = 0; r < 8; ++r) {
                    float v = fmaxf(c[r], 0.f);
                    out_p[(e0 + mrow + r) * 128 + (n2 - 128)] = v;
                }
            } else {                                     // new_o -> pooled[o]
#pragma unroll
                for (int r = 0; r < 8; ++r) {
                    float v = fmaxf(c[r], 0.f);
                    atomicAdd(&pooled[(long)Eo[mrow + r] * 128 + (n2 - 256)], v);
                }
            }
        }
    }
}

// ------------------- final object MLP: 128 -> 256 -> 128 --------------------
__global__ __launch_bounds__(256) void k_obj(const float* __restrict__ pooled,
                                             const float* __restrict__ counts,
                                             const float* __restrict__ W3,
                                             const float* __restrict__ b3,
                                             const float* __restrict__ W4,
                                             const float* __restrict__ b4,
                                             float* __restrict__ out) {
    __shared__ float xs[128];
    __shared__ float hs[256];
    int i = blockIdx.x, t = threadIdx.x;
    if (t < 128) {
        float cn = counts[i];
        cn = cn < 1.f ? 1.f : cn;
        xs[t] = pooled[i * 128 + t] / cn;
    }
    __syncthreads();
    float acc = b3[t];
    for (int k = 0; k < 128; ++k) acc = fmaf(xs[k], W3[k * 256 + t], acc);
    hs[t] = acc > 0.f ? acc : 0.f;
    __syncthreads();
    if (t < 128) {
        float a2 = b4[t];
        for (int k = 0; k < 256; ++k) a2 = fmaf(hs[k], W4[k * 128 + t], a2);
        out[i * 128 + t] = a2 > 0.f ? a2 : 0.f;
    }
}

// ---------------------------------------------------------------------------
extern "C" void kernel_launch(void* const* d_in, const int* in_sizes, int n_in,
                              void* d_out, int out_size, void* d_ws, size_t ws_size,
                              hipStream_t stream) {
    const float* obj  = (const float*)d_in[0];
    const float* pred = (const float*)d_in[1];
    const int*   edges= (const int*)  d_in[2];
    const float* W    = (const float*)d_in[3];
    const float* a    = (const float*)d_in[4];
    const float* W1   = (const float*)d_in[5];
    const float* b1   = (const float*)d_in[6];
    const float* W2   = (const float*)d_in[7];
    const float* b2   = (const float*)d_in[8];
    const float* W3   = (const float*)d_in[9];
    const float* b3   = (const float*)d_in[10];
    const float* W4   = (const float*)d_in[11];
    const float* b4   = (const float*)d_in[12];
    float* out = (float*)d_out;

    float* ws     = (float*)d_ws;
    float* Wh     = ws;                        // 512*128
    float* f      = Wh + O_N * D;              // 512
    float* g      = f + O_N;                   // 512
    float* hp     = g + O_N;                   // 512*128
    float* pooled = hp + O_N * D;              // 512*128
    float* counts = pooled + O_N * D;          // 512
    __bf16* bpack1 = (__bf16*)(counts + O_N);  // 128*384 bf16, 16B aligned
    __bf16* bpack2 = bpack1 + 128 * 384;       // 384*128 bf16

    k_prep  <<<384, 256, 0, stream>>>(W1, W2, bpack1, bpack2);
    k_wh    <<<O_N, 128, 0, stream>>>(obj, W, a, Wh, f, g);
    k_att   <<<O_N, 128, 0, stream>>>(Wh, f, g, hp);
    k_zero  <<<(O_N * D + O_N + 255) / 256, 256, 0, stream>>>(pooled, O_N * D + O_N);
    k_counts<<<(T_EDGES + 255) / 256, 256, 0, stream>>>(edges, counts, T_EDGES);
    k_edge  <<<T_EDGES / TM, 256, 0, stream>>>(hp, pred, edges, bpack1, b1, bpack2, b2,
                                               pooled, out + O_N * D);
    k_obj   <<<O_N, 256, 0, stream>>>(pooled, counts, W3, b3, W4, b4, out);
}
